// PairElementEnergyScattering_56564719289128
// MI455X (gfx1250) — matrix operand order repaired and checked by reading, not compiled, not run on hardware
//
#include <hip/hip_runtime.h>

// ---------------------------------------------------------------------------
// Fused PairElementEnergyScattering MLP for MI455X (gfx1250, wave32, WMMA).
//
// Roofline: mandatory HBM traffic = 52 MB output (~2.3us @ 23.3 TB/s); the
// ~30 GFLOP of layer-2/3 GEMMs run on v_wmma_f32_16x16x32_f16 (f32 accum);
// he/W2/W3 stay L2-resident (192 MB L2). Time-dominant term is the
// 2 x 104M elementwise silu evals: silu(x) = h + h*tanh(h), h = x/2, costs
// v_fma (add+halve fused via pre-halved operands) + v_tanh_f32 (1 TRANS)
// + v_fma_mixlo_f16 (fma+f16-cvt fused) = 2 VALU + 1 TRANS per element.
// ---------------------------------------------------------------------------

typedef __attribute__((ext_vector_type(16))) _Float16 v16h;
typedef __attribute__((ext_vector_type(8)))  _Float16 v8h;
typedef __attribute__((ext_vector_type(8)))  float    v8f;

#define ZEMB 64
#define EDIM 32
#define HIDN 128
#define OUTD 16
#define W_LD 136   // padded LDS row stride (halves) for W2T/W3T: 272 B = 17*16 B
#define S_LD 40    // padded LDS row stride (halves) for per-wave scratch

__device__ __forceinline__ float tanh_fast(float x) {
#if __has_builtin(__builtin_amdgcn_tanhf)
  return __builtin_amdgcn_tanhf(x);
#elif __has_builtin(__builtin_amdgcn_tanh_f32)
  return __builtin_amdgcn_tanh_f32(x);
#else
  float r;
  asm("v_tanh_f32 %0, %1\n\tv_nop" : "=v"(r) : "v"(x));
  return r;
#endif
}

// silu(x) expressed from h = x/2:  silu = h + h*tanh(h).
__device__ __forceinline__ _Float16 silu_from_h(float h) {
  const float t = tanh_fast(h);
  return (_Float16)__builtin_fmaf(h, t, h);   // -> v_fma_mixlo/hi_f16
}

// ---- prep: hp[bp][n] = b1[n] + ej@Wj + ek@Wk  (128 threads/block) ----------
__global__ void pees_hp_kernel(const int* __restrict__ z_j, const int* __restrict__ z_k,
                               const float* __restrict__ z_emb, const float* __restrict__ W1,
                               const float* __restrict__ b1, float* __restrict__ hp) {
  const int bp = blockIdx.x;
  const int n  = threadIdx.x;
  const float* ej = z_emb + (size_t)z_j[bp] * ZEMB;
  const float* ek = z_emb + (size_t)z_k[bp] * ZEMB;
  float acc = b1[n];
  #pragma unroll 8
  for (int k = 0; k < ZEMB; ++k) acc += ej[k] * W1[k * HIDN + n];
  #pragma unroll 8
  for (int k = 0; k < ZEMB; ++k) acc += ek[k] * W1[(ZEMB + k) * HIDN + n];
  hp[(size_t)bp * HIDN + n] = acc;
}

// ---- prep: he[e][n] = e_feat[e] @ We  (128 threads/block) ------------------
__global__ void pees_he_kernel(const float* __restrict__ e_feat, const float* __restrict__ W1,
                               float* __restrict__ he) {
  const int e = blockIdx.x;
  const int n = threadIdx.x;
  float acc = 0.0f;
  #pragma unroll
  for (int k = 0; k < EDIM; ++k)
    acc += e_feat[e * EDIM + k] * W1[(2 * ZEMB + k) * HIDN + n];
  he[(size_t)e * HIDN + n] = acc;
}

// Load a 16-half B fragment from LDS as 2x 16-byte chunks at p and p+16 (16-B aligned).
__device__ __forceinline__ v16h lds_frag16(const _Float16* p) {
  v8h lo = *(const v8h*)p;
  v8h hi = *(const v8h*)(p + 16);
  v16h r;
  #pragma unroll
  for (int j = 0; j < 8; ++j) { r[j] = lo[j]; r[8 + j] = hi[j]; }
  return r;
}

// ---- main fused kernel: 1 block per (b,p); 8 waves; 16 e-rows per WMMA tile -
__global__ __launch_bounds__(256) void pees_main_kernel(
    const float* __restrict__ hp, const float* __restrict__ he,
    const float* __restrict__ W2, const float* __restrict__ b2,
    const float* __restrict__ W3, const float* __restrict__ b3,
    float* __restrict__ out, int nE, int ntiles)
{
  __shared__ _Float16 sW2T[HIDN * W_LD];     // W2^T f16: [n][k]
  __shared__ _Float16 sW3T[OUTD * W_LD];     // W3^T f16: [n][k]
  __shared__ _Float16 sScr[8][16 * S_LD];    // per-wave 16x32 transpose buffer
  __shared__ float    sHp[HIDN];             // 0.5 * hp row (pre-halved for silu fma)
  __shared__ float    sB2[HIDN];             // 0.5 * b2     (pre-halved for silu fma)
  __shared__ float    sB3[OUTD];

  const int t  = threadIdx.x;
  const int bp = blockIdx.x;

  // Stage f16 transposed weights + pre-halved silu operands into LDS.
  for (int i = t; i < HIDN * HIDN; i += 256) {
    const int k = i >> 7, n = i & 127;       // coalesced row-major read of W2
    sW2T[n * W_LD + k] = (_Float16)W2[i];
  }
  for (int i = t; i < HIDN * OUTD; i += 256) {
    const int k = i >> 4, n = i & 15;
    sW3T[n * W_LD + k] = (_Float16)W3[i];
  }
  if (t < HIDN) { sHp[t] = 0.5f * hp[(size_t)bp * HIDN + t]; sB2[t] = 0.5f * b2[t]; }
  if (t < OUTD) sB3[t] = b3[t];
  __syncthreads();

  const int wave = t >> 5;
  const int lane = t & 31;
  const int m    = lane & 15;   // A row within tile / C-N index
  const int half = lane >> 4;   // K-half selector per ISA A/B layouts

  _Float16* scr = &sScr[wave][0];

  // Tile-invariant operands hoisted out of the e-tile loop.
  v16h B3[4];
  #pragma unroll
  for (int ks = 0; ks < 4; ++ks)
    B3[ks] = lds_frag16(&sW3T[m * W_LD + ks * 32 + half * 8]);
  float b2h[8];
  #pragma unroll
  for (int nt = 0; nt < 8; ++nt) b2h[nt] = sB2[nt * 16 + m];  // 0.5*b2
  const float b3v = sB3[m];

  for (int et = wave; et < ntiles; et += 8) {
    const int e0 = et * 16;
    int erow = e0 + m; if (erow >= nE) erow = nE - 1;
    const float* heR = he + (size_t)erow * HIDN;

    if (e0 + 128 < nE)   // warm L2->WGP path for a future tile
      __builtin_prefetch(he + (size_t)(e0 + 128) * HIDN, 0, 1);

    // Phase 1: issue ALL he loads for this tile up front (16x b128, one wait).
    float4 G[16];
    #pragma unroll
    for (int ks = 0; ks < 4; ++ks) {
      const int kb = ks * 32 + half * 8;
      G[4 * ks + 0] = *(const float4*)(heR + kb);
      G[4 * ks + 1] = *(const float4*)(heR + kb + 4);
      G[4 * ks + 2] = *(const float4*)(heR + kb + 16);
      G[4 * ks + 3] = *(const float4*)(heR + kb + 20);
    }

    // Phase 2: layer-2 A fragments (16x32 f16, ISA layout) with fused silu.
    // h = fma(he, 0.5, 0.5*hp); A = cvt_f16(fma(h, tanh(h), h)).
    v16h A[4];
    #pragma unroll
    for (int ks = 0; ks < 4; ++ks) {
      const int kb = ks * 32 + half * 8;
      const float* gl = (const float*)&G[4 * ks];      // he[kb .. kb+7]
      const float* gh = (const float*)&G[4 * ks + 2];  // he[kb+16 .. kb+23]
      const float* hp05 = sHp + kb;
      #pragma unroll
      for (int j = 0; j < 8; ++j) {
        A[ks][j]     = silu_from_h(__builtin_fmaf(gl[j], 0.5f, hp05[j]));
        A[ks][8 + j] = silu_from_h(__builtin_fmaf(gh[j], 0.5f, hp05[16 + j]));
      }
    }

    // Layer 2 (16x128x128) interleaved with layer 3 (16x128x16):
    // each ks3 consumes the two layer-2 N-tiles that form layer-3's K-slice.
    v8f acc3 = {};
    #pragma unroll
    for (int ks3 = 0; ks3 < 4; ++ks3) {
      #pragma unroll
      for (int h = 0; h < 2; ++h) {
        const int nt = ks3 * 2 + h;          // layer-2 N tile index
        v16h Bf[4];
        #pragma unroll
        for (int kk = 0; kk < 4; ++kk)
          Bf[kk] = lds_frag16(&sW2T[(nt * 16 + m) * W_LD + kk * 32 + half * 8]);
        v8f acc = {};
        #pragma unroll
        for (int kk = 0; kk < 4; ++kk)
          acc = __builtin_amdgcn_wmma_f32_16x16x32_f16(
              false, A[kk], false, Bf[kk], (short)0, acc, false, false);
        // bias + silu; transpose N-major D -> M-major A via per-wave LDS.
        #pragma unroll
        for (int r = 0; r < 8; ++r) {
          const int mm = r + half * 8;       // D frag: VGPR r is row r (+8 for hi lanes)
          scr[mm * S_LD + h * 16 + m] =
              silu_from_h(__builtin_fmaf(acc[r], 0.5f, b2h[nt]));
        }
      }
      // Layer-3 A fragment for K-slice ks3 (scratch columns 0..31).
      const int kb = half * 8;
      const v8h lo = *(const v8h*)&scr[m * S_LD + kb];
      const v8h hi = *(const v8h*)&scr[m * S_LD + kb + 16];
      v16h A3;
      #pragma unroll
      for (int j = 0; j < 8; ++j) { A3[j] = lo[j]; A3[8 + j] = hi[j]; }
      acc3 = __builtin_amdgcn_wmma_f32_16x16x32_f16(
          false, A3, false, B3[ks3], (short)0, acc3, false, false);
    }

    // out[bp][e0+mm][m] = acc3 + b3 (D frag is N-major: lane N = m, VGPR r -> row)
    #pragma unroll
    for (int r = 0; r < 8; ++r) {
      const int mm = r + half * 8;
      if (e0 + mm < nE)
        out[((size_t)bp * nE + (e0 + mm)) * OUTD + m] = acc3[r] + b3v;
    }
  }
}

extern "C" void kernel_launch(void* const* d_in, const int* in_sizes, int n_in,
                              void* d_out, int out_size, void* d_ws, size_t ws_size,
                              hipStream_t stream) {
  const int*   z_j    = (const int*)d_in[0];
  const int*   z_k    = (const int*)d_in[1];
  const float* e_feat = (const float*)d_in[2];
  const float* z_emb  = (const float*)d_in[3];
  const float* W1     = (const float*)d_in[4];
  const float* b1     = (const float*)d_in[5];
  const float* W2     = (const float*)d_in[6];
  const float* b2     = (const float*)d_in[7];
  const float* W3     = (const float*)d_in[8];
  const float* b3     = (const float*)d_in[9];
  float*       out    = (float*)d_out;

  const int BP = in_sizes[0];          // B*P = 2048
  const int nE = in_sizes[2] / EDIM;   // 400
  const int ntiles = (nE + 15) / 16;   // 25

  float* hp = (float*)d_ws;
  float* he = hp + (size_t)BP * HIDN;

  pees_hp_kernel<<<BP, HIDN, 0, stream>>>(z_j, z_k, z_emb, W1, b1, hp);
  pees_he_kernel<<<nE, HIDN, 0, stream>>>(e_feat, W1, he);
  pees_main_kernel<<<BP, 256, 0, stream>>>(hp, he, W2, b2, W3, b3, out, nE, ntiles);
}
